// UltimateV6_48979807043619
// MI455X (gfx1250) — compile-verified, matrix operand order checked by reading, not run on hardware
//
#include <hip/hip_runtime.h>
#include <hip/hip_bf16.h>
#include <math.h>

// ---------------- CDNA5 (gfx1250) fused dynamic-MoE-MLP forward ----------------
// wave32, WMMA f16->f32. See analysis: compute-bound (~275 GFLOP vs 85MB HBM).

#define NLAYERS 4
#define DIMM    128
#define NBLK    8
#define FINN    32
#define RPW     32   // rows per workgroup

typedef __attribute__((ext_vector_type(16))) _Float16 v16h;
typedef __attribute__((ext_vector_type(8)))  float    v8f;

typedef union { v16h h; unsigned int u[8]; uint4 q[2]; } Frag;

// ISA 16-bit A/B fragment K mapping (CDNA5 ISA, 16-bit A-Matrix 16x32):
// VGPR r (0..3): K = 2r + 8*hi + p ; VGPR r (4..7): K = 16 + 2(r-4) + 8*hi + p
__device__ __forceinline__ int kfrag(int r, int hi, int p) {
  return ((r & 3) << 1) + ((r >> 2) << 4) + (hi << 3) + p;
}
__device__ __forceinline__ float sigmoid_f(float x) { return 1.0f / (1.0f + expf(-x)); }
__device__ __forceinline__ float gelu_f(float x) {
  return 0.5f * x * (1.0f + erff(x * 0.70710678118654752440f));
}

// ---------------------------------------------------------------------------
// Phase B: evolve row 0 in fp32, pick top-2 blocks per layer (jax tie-break:
// lower index first), write 8 indices to ws. Single workgroup of 128 threads.
// ---------------------------------------------------------------------------
__global__ void moe_select_kernel(
    const float* __restrict__ x, const float* __restrict__ proj_w,
    const float* __restrict__ proj_b, const float* __restrict__ pos,
    const float* __restrict__ sel_w, const float* __restrict__ sel_b,
    const float* __restrict__ ln_s, const float* __restrict__ ln_b,
    const float* __restrict__ w1, const float* __restrict__ b1,
    const float* __restrict__ w2, const float* __restrict__ b2,
    const float* __restrict__ gate, const float* __restrict__ tf,
    int* __restrict__ sel_out)
{
  __shared__ float h0[DIMM], hn[DIMM], g[2 * DIMM], scr[DIMM], sc[NBLK];
  __shared__ int seli[2];
  const int t = threadIdx.x;  // 0..127

  // h0 = x[0] @ proj_w + proj_b + pos[0,0,:]
  {
    float acc = proj_b[t] + pos[t];
    for (int k = 0; k < FINN; ++k) acc += x[k] * proj_w[k * DIMM + t];
    h0[t] = acc;
  }
  __syncthreads();

  for (int l = 0; l < NLAYERS; ++l) {
    if (t < NBLK) {
      float s = sel_b[l * NBLK + t];
      for (int d = 0; d < DIMM; ++d) s += h0[d] * sel_w[(l * DIMM + d) * NBLK + t];
      sc[t] = 0.6f * sigmoid_f(s) + 0.2f;  // 0.4 * priority(0.5)
    }
    __syncthreads();
    if (t == 0) {
      int i0 = 0;
      for (int j = 1; j < NBLK; ++j) if (sc[j] > sc[i0]) i0 = j;
      int i1 = (i0 == 0) ? 1 : 0;
      for (int j = 0; j < NBLK; ++j) if (j != i0 && sc[j] > sc[i1]) i1 = j;
      seli[0] = i0; seli[1] = i1;
      sel_out[l * 2 + 0] = i0; sel_out[l * 2 + 1] = i1;
    }
    __syncthreads();

    for (int kk = 0; kk < 2; ++kk) {
      const int blk = l * NBLK + seli[kk];
      // mean
      scr[t] = h0[t]; __syncthreads();
      for (int s = 64; s >= 1; s >>= 1) { if (t < s) scr[t] += scr[t + s]; __syncthreads(); }
      const float mu = scr[0] * (1.0f / DIMM);
      __syncthreads();
      // var
      { float v = h0[t] - mu; scr[t] = v * v; } __syncthreads();
      for (int s = 64; s >= 1; s >>= 1) { if (t < s) scr[t] += scr[t + s]; __syncthreads(); }
      const float rstd = rsqrtf(scr[0] * (1.0f / DIMM) + 1e-5f);
      __syncthreads();
      hn[t] = (h0[t] - mu) * rstd * ln_s[blk * DIMM + t] + ln_b[blk * DIMM + t];
      __syncthreads();
      // z1 = gelu(hn @ w1 + b1)
      for (int jj = t; jj < 2 * DIMM; jj += DIMM) {
        float acc = b1[blk * 2 * DIMM + jj];
        for (int d = 0; d < DIMM; ++d) acc += hn[d] * w1[((size_t)blk * DIMM + d) * 2 * DIMM + jj];
        g[jj] = gelu_f(acc);
      }
      __syncthreads();
      // z2 = z1 @ w2 + b2 ; torsion gate ; 0.5 residual
      {
        float acc = b2[blk * DIMM + t];
        for (int j = 0; j < 2 * DIMM; ++j) acc += g[j] * w2[((size_t)blk * 2 * DIMM + j) * DIMM + t];
        const float te = sigmoid_f(gate[blk * DIMM + t]) * tf[t];
        h0[t] += 0.5f * acc * (1.0f + te);
      }
      __syncthreads();
    }
  }
}

// ---------------------------------------------------------------------------
// Phase A: convert the 8 selected blocks' w1/w2 to f16, pre-swizzled into
// WMMA B-fragment order: dst[((kt*NT + nt)*32 + lane)*16 + j], lane<->N, j<->K.
// 32768 threads total (128 blocks x 256).
// ---------------------------------------------------------------------------
__global__ void moe_convert_kernel(
    const float* __restrict__ w1, const float* __restrict__ w2,
    const int* __restrict__ sel,
    _Float16* __restrict__ w1h, _Float16* __restrict__ w2h)
{
  const int gid  = blockIdx.x * 256 + threadIdx.x;
  const int lane = gid & 31;
  const int f    = gid >> 5;        // 0..1023
  const int app  = f >> 7;          // 0..7
  const int fi   = f & 127;
  const int l    = app >> 1;
  const int blk  = l * NBLK + sel[app];
  const int hi   = (lane >> 4) & 1;
  const int n    = lane & 15;

  if (fi < 64) {                    // w1: 128x256, KT=4, NT=16
    const int kt = fi >> 4, nt = fi & 15;
    _Float16* dst = w1h + (size_t)app * 32768 + ((size_t)(kt * 16 + nt) * 32 + lane) * 16;
    for (int j = 0; j < 16; ++j) {
      const int K = kt * 32 + kfrag(j >> 1, hi, j & 1);
      const int N = nt * 16 + n;
      dst[j] = (_Float16)w1[((size_t)blk * DIMM + K) * (2 * DIMM) + N];
    }
  } else {                          // w2: 256x128, KT=8, NT=8
    const int f2 = fi - 64;
    const int kt = f2 >> 3, nt = f2 & 7;
    _Float16* dst = w2h + (size_t)app * 32768 + ((size_t)(kt * 8 + nt) * 32 + lane) * 16;
    for (int j = 0; j < 16; ++j) {
      const int K = kt * 32 + kfrag(j >> 1, hi, j & 1);
      const int N = nt * 16 + n;
      dst[j] = (_Float16)w2[((size_t)blk * 2 * DIMM + K) * DIMM + N];
    }
  }
}

// te[app][d] = sigmoid(gate[blk,d]) * torsion_field[d]
__global__ void moe_te_kernel(const float* __restrict__ gate, const float* __restrict__ tf,
                              const int* __restrict__ sel, float* __restrict__ te)
{
  const int idx = blockIdx.x * 256 + threadIdx.x;
  if (idx >= 8 * DIMM) return;
  const int app = idx >> 7, d = idx & 127;
  const int l = app >> 1;
  const int blk = l * NBLK + sel[app];
  te[idx] = sigmoid_f(gate[blk * DIMM + d]) * tf[d];
}

// ---------------------------------------------------------------------------
// Phase C: fused forward. 256 threads = 8 waves, 32 rows per workgroup.
// h stays resident in LDS across all 8 block applications; GEMMs via WMMA.
// ---------------------------------------------------------------------------
__global__ void __launch_bounds__(256)
moe_forward_kernel(
    const float* __restrict__ x, const float* __restrict__ proj_w,
    const float* __restrict__ proj_b, const float* __restrict__ pos,
    const float* __restrict__ b1, const float* __restrict__ b2,
    const float* __restrict__ ln_s, const float* __restrict__ ln_b,
    const float* __restrict__ cls_w, const float* __restrict__ cls_b,
    const int* __restrict__ sel, const float* __restrict__ te,
    const _Float16* __restrict__ w1h, const _Float16* __restrict__ w2h,
    float* __restrict__ out, int nrows)
{
  __shared__ float    h32[RPW * DIMM];          // fp32 hidden state (resident)
  __shared__ _Float16 hn16[RPW * DIMM];         // layernormed, f16 (GEMM1 A)
  __shared__ _Float16 z16[RPW * 2 * DIMM];      // gelu output, f16 (GEMM2 A)
  __shared__ float    xs[RPW * FINN];
  __shared__ float    pb1[2 * DIMM], pb2[DIMM], pte[DIMM], plns[DIMM], plnb[DIMM];
  __shared__ float    red[RPW * 8], means[RPW], rstd[RPW];

  const int tid  = threadIdx.x;
  const int wave = tid >> 5;
  const int lane = tid & 31;
  const int m    = lane & 15;
  const int hi   = (lane >> 4) & 1;
  const int rowbase = blockIdx.x * RPW;

  // ---- load x tile, fp32 input projection (kept fp32: tiny FLOP share) ----
  for (int idx = tid; idx < RPW * FINN; idx += 256)
    xs[idx] = x[(size_t)(rowbase + (idx >> 5)) * FINN + (idx & 31)];
  __syncthreads();
  {
    const int r = tid >> 3, dg = tid & 7;
    for (int dd = 0; dd < 16; ++dd) {
      const int d = dg * 16 + dd;
      float acc = proj_b[d] + pos[d];
      for (int k = 0; k < FINN; ++k) acc += xs[r * FINN + k] * proj_w[k * DIMM + d];
      h32[r * DIMM + d] = acc;
    }
  }

  // ---- 8 block applications ----
  for (int app = 0; app < 8; ++app) {
    const int l   = app >> 1;
    const int blk = l * NBLK + sel[app];
    const _Float16* w1a = w1h + (size_t)app * 32768;
    const _Float16* w2a = w2h + (size_t)app * 32768;

    // prefetch this app's swizzled weights (64 KB each, 256 B per thread)
    __builtin_prefetch((const char*)w1a + tid * 256, 0, 0);
    __builtin_prefetch((const char*)w2a + tid * 256, 0, 0);

    // stage per-block params
    if (tid < DIMM) {
      plns[tid] = ln_s[blk * DIMM + tid];
      plnb[tid] = ln_b[blk * DIMM + tid];
      pb2[tid]  = b2[blk * DIMM + tid];
      pte[tid]  = te[app * DIMM + tid];
    }
    pb1[tid] = b1[blk * 2 * DIMM + tid];
    __syncthreads();

    // ---- LayerNorm over DIM=128 ----
    {
      const int r = tid >> 3, seg = tid & 7;
      float s = 0.f;
      for (int dd = 0; dd < 16; ++dd) s += h32[r * DIMM + seg * 16 + dd];
      red[tid] = s;
      __syncthreads();
      if (tid < RPW) {
        float mm = 0.f;
        for (int i = 0; i < 8; ++i) mm += red[tid * 8 + i];
        means[tid] = mm * (1.0f / DIMM);
      }
      __syncthreads();
      const float mu = means[r];
      s = 0.f;
      for (int dd = 0; dd < 16; ++dd) {
        const float v = h32[r * DIMM + seg * 16 + dd] - mu;
        s += v * v;
      }
      red[tid] = s;
      __syncthreads();
      if (tid < RPW) {
        float vv = 0.f;
        for (int i = 0; i < 8; ++i) vv += red[tid * 8 + i];
        rstd[tid] = rsqrtf(vv * (1.0f / DIMM) + 1e-5f);
      }
      __syncthreads();
      for (int idx = tid; idx < RPW * DIMM; idx += 256) {
        const int rr = idx >> 7, d = idx & 127;
        hn16[idx] = (_Float16)((h32[idx] - means[rr]) * rstd[rr] * plns[d] + plnb[d]);
      }
    }
    __syncthreads();

    // ---- GEMM1: [32x128] @ [128x256] -> gelu -> z16, WMMA f16 ----
    {
      const int rt  = (wave >= 4) ? 1 : 0;
      const int ntb = (wave & 3) * 4;
      v8f acc[4];
      for (int i = 0; i < 4; ++i) acc[i] = v8f{0.f, 0.f, 0.f, 0.f, 0.f, 0.f, 0.f, 0.f};
      for (int kt = 0; kt < 4; ++kt) {
        Frag A;
        #pragma unroll
        for (int r8 = 0; r8 < 8; ++r8) {
          const int K = kt * 32 + kfrag(r8, hi, 0);  // even -> 4B aligned pair
          A.u[r8] = *(const unsigned int*)&hn16[(rt * 16 + m) * DIMM + K];
        }
        #pragma unroll
        for (int i = 0; i < 4; ++i) {
          const int nt = ntb + i;
          Frag B;
          const uint4* q = (const uint4*)(w1a + ((size_t)(kt * 16 + nt) * 32 + lane) * 16);
          B.q[0] = q[0]; B.q[1] = q[1];
          acc[i] = __builtin_amdgcn_wmma_f32_16x16x32_f16(
              false, A.h, false, B.h, (short)0, acc[i], false, false);
        }
      }
      #pragma unroll
      for (int i = 0; i < 4; ++i) {
        const int col = (ntb + i) * 16 + m;
        const float bc = pb1[col];
        #pragma unroll
        for (int r8 = 0; r8 < 8; ++r8) {
          const int row = rt * 16 + r8 + hi * 8;
          z16[row * (2 * DIMM) + col] = (_Float16)gelu_f(acc[i][r8] + bc);
        }
      }
    }
    __syncthreads();

    // ---- GEMM2: [32x256] @ [256x128] -> torsion scale -> 0.5 residual ----
    {
      const int rt  = (wave >= 4) ? 1 : 0;
      const int ntb = (wave & 3) * 2;
      v8f acc[2];
      for (int i = 0; i < 2; ++i) acc[i] = v8f{0.f, 0.f, 0.f, 0.f, 0.f, 0.f, 0.f, 0.f};
      for (int kt = 0; kt < 8; ++kt) {
        Frag A;
        #pragma unroll
        for (int r8 = 0; r8 < 8; ++r8) {
          const int K = kt * 32 + kfrag(r8, hi, 0);
          A.u[r8] = *(const unsigned int*)&z16[(rt * 16 + m) * (2 * DIMM) + K];
        }
        #pragma unroll
        for (int i = 0; i < 2; ++i) {
          const int nt = ntb + i;
          Frag B;
          const uint4* q = (const uint4*)(w2a + ((size_t)(kt * 8 + nt) * 32 + lane) * 16);
          B.q[0] = q[0]; B.q[1] = q[1];
          acc[i] = __builtin_amdgcn_wmma_f32_16x16x32_f16(
              false, A.h, false, B.h, (short)0, acc[i], false, false);
        }
      }
      #pragma unroll
      for (int i = 0; i < 2; ++i) {
        const int col = (ntb + i) * 16 + m;
        const float sc2 = 1.0f + pte[col];
        const float b2c = pb2[col];
        #pragma unroll
        for (int r8 = 0; r8 < 8; ++r8) {
          const int row = rt * 16 + r8 + hi * 8;
          const int hidx = row * DIMM + col;
          h32[hidx] = h32[hidx] + 0.5f * ((acc[i][r8] + b2c) * sc2);
        }
      }
    }
    __syncthreads();
  }

  // ---- classifier + h writeout ----
  if (tid < 64) {
    const int r = tid >> 1, c = tid & 1;
    float acc = cls_b[c];
    for (int d = 0; d < DIMM; ++d) acc += h32[r * DIMM + d] * cls_w[d * 2 + c];
    out[(size_t)(rowbase + r) * 2 + c] = acc;
  }
  float* outH = out + (size_t)nrows * 2;
  for (int idx = tid; idx < RPW * DIMM; idx += 256) {
    const int r = idx >> 7, d = idx & 127;
    outH[(size_t)(rowbase + r) * DIMM + d] = h32[idx];
  }
}

// ---------------------------------------------------------------------------
extern "C" void kernel_launch(void* const* d_in, const int* in_sizes, int n_in,
                              void* d_out, int out_size, void* d_ws, size_t ws_size,
                              hipStream_t stream) {
  const float* x      = (const float*)d_in[0];
  const float* proj_w = (const float*)d_in[1];
  const float* proj_b = (const float*)d_in[2];
  const float* pos    = (const float*)d_in[3];
  const float* sel_w  = (const float*)d_in[4];
  const float* sel_b  = (const float*)d_in[5];
  const float* ln_s   = (const float*)d_in[6];
  const float* ln_b   = (const float*)d_in[7];
  const float* w1     = (const float*)d_in[8];
  const float* b1     = (const float*)d_in[9];
  const float* w2     = (const float*)d_in[10];
  const float* b2     = (const float*)d_in[11];
  const float* gate   = (const float*)d_in[12];
  const float* cls_w  = (const float*)d_in[13];
  const float* cls_b  = (const float*)d_in[14];
  const float* tf     = (const float*)d_in[15];
  float* out = (float*)d_out;

  char* ws = (char*)d_ws;
  int*       sel = (int*)ws;                          // 8 ints
  float*     te  = (float*)(ws + 1024);               // 8*128 f32
  _Float16*  w1h = (_Float16*)(ws + 8192);            // 8*32768 f16 (512 KB)
  _Float16*  w2h = (_Float16*)(ws + 8192 + (size_t)8 * 32768 * 2);  // 512 KB

  const int nrows = in_sizes[0] / FINN;               // 131072

  moe_select_kernel<<<1, 128, 0, stream>>>(x, proj_w, proj_b, pos, sel_w, sel_b,
                                           ln_s, ln_b, w1, b1, w2, b2, gate, tf, sel);
  moe_convert_kernel<<<128, 256, 0, stream>>>(w1, w2, sel, w1h, w2h);
  moe_te_kernel<<<4, 256, 0, stream>>>(gate, tf, sel, te);
  moe_forward_kernel<<<nrows / RPW, 256, 0, stream>>>(
      x, proj_w, proj_b, pos, b1, b2, ln_s, ln_b, cls_w, cls_b,
      sel, te, w1h, w2h, out, nrows);
}
